// MessageBlock_16784732193373
// MI455X (gfx1250) — matrix operand order repaired and checked by reading, not compiled, not run on hardware
//
#include <hip/hip_runtime.h>
#include <hip/hip_bf16.h>

// MPNN block: NNConv(edge-conditioned) + mean-aggr + CELU + GRU step + residual ReLU.
//
// Key optimization: never materialize W[E,32,32] (1.02 GB of HBM traffic).
//   msg[e] = sum_{d=0}^{13} u[e,d] * (x[src[e]] @ A_d),  u = [edge_attr, 1]
// computed as a K=448 bf16 WMMA chain per 16-edge tile, with the per-edge
// scale u[e,d] folded into the A operand (A rows are lane-local on CDNA5).
// Weight tables are staged to LDS via the Tensor Data Mover (edge kernel)
// and async global->LDS loads (node kernel).

#define N_NODES 100000
#define N_EDGES 250000
#define DIN 32
#define DOUT 32
#define DE 13
#define DU 14            // DE + 1 (implicit bias column)
#define ETILES (N_EDGES / 16)   // 15625 exactly
#define NTILES (N_NODES / 16)   // 6250 exactly

typedef __attribute__((ext_vector_type(16))) __bf16 v16bf;
typedef __attribute__((ext_vector_type(8)))  float  v8f;
typedef __attribute__((ext_vector_type(4)))  unsigned int v4u;
typedef __attribute__((ext_vector_type(8)))  unsigned int v8u;

// workspace layout (bytes)
#define OFF_AGG   0u
#define OFF_CNT   (N_NODES * DOUT * 4u)            // 12,800,000
#define OFF_BMSG  (OFF_CNT + N_NODES * 4u)         // 13,200,000 (32B aligned)
#define OFF_BNODE (OFF_BMSG + 28u * 512u * 2u)     // 13,228,672

// ---------------------------------------------------------------------------
// K0: repack all weight matrices into bf16 WMMA-B layout (K=32, N=16 tiles).
// B layout (16-bit, 32x16): lane L holds column n = L%16; lanes 0-15 hold
// K=0..7,16..23, lanes 16-31 hold K=8..15,24..31; v16bf element j <-> K = 8*(L/16) + j + (j>=8 ? 8 : 0).
// ---------------------------------------------------------------------------
__global__ void pack_weights(const float* __restrict__ nn_w,
                             const float* __restrict__ nn_b,
                             const float* __restrict__ root,
                             const float* __restrict__ w_ih,
                             const float* __restrict__ w_hh,
                             __bf16* __restrict__ bmsg,
                             __bf16* __restrict__ bnode) {
  int idx = blockIdx.x * blockDim.x + threadIdx.x;
  int stride = gridDim.x * blockDim.x;
  for (int i = idx; i < 28 * 512; i += stride) {
    int tile = i >> 9, u = i & 511;
    int d = tile >> 1, nt = tile & 1;
    int L = u >> 4, j = u & 15;
    int n = nt * 16 + (L & 15);
    int k = ((L < 16) ? 0 : 8) + ((j < 8) ? j : 8 + j);
    float v = (d < DE) ? nn_w[(k * DOUT + n) * DE + d] : nn_b[k * DOUT + n];
    bmsg[i] = (__bf16)v;
  }
  for (int i = idx; i < 14 * 512; i += stride) {
    int tile = i >> 9, u = i & 511;
    int L = u >> 4, j = u & 15;
    int nl = L & 15;
    int k = ((L < 16) ? 0 : 8) + ((j < 8) ? j : 8 + j);
    float v;
    if (tile < 2)      v = root[k * DOUT + tile * 16 + nl];         // x @ root
    else if (tile < 8) v = w_ih[((tile - 2) * 16 + nl) * DIN + k];  // c @ w_ih^T
    else               v = w_hh[((tile - 8) * 16 + nl) * DOUT + k]; // h @ w_hh^T
    bnode[i] = (__bf16)v;
  }
}

// ---------------------------------------------------------------------------
// K1: zero the aggregation buffers (agg[N,32] and cnt[N] are contiguous).
// ---------------------------------------------------------------------------
__global__ void zero_ws(float* __restrict__ p) {
  int i = blockIdx.x * blockDim.x + threadIdx.x;
  if (i < N_NODES * (DOUT + 1)) p[i] = 0.0f;
}

// ---------------------------------------------------------------------------
// K2: edge kernel. One wave32 per 16-edge tile; 8 tiles per 256-thread block.
// Weight table (28 KB) staged to LDS with a single TDM descriptor:
//   D# group0: count=1 | lds_addr | global_addr | type=2
//   D# group1: data_size=4B, tile 7168x1 dwords, dim0_stride=7168
// ---------------------------------------------------------------------------
__global__ __launch_bounds__(256) void edge_kernel(
    const float* __restrict__ x, const float* __restrict__ edge_attr,
    const int* __restrict__ edge_index, const __bf16* __restrict__ bmsg,
    float* __restrict__ agg, float* __restrict__ cnt) {
  __shared__ __align__(32) __bf16 Bsm[28 * 512];

  if (threadIdx.x < 32) {  // wave 0 issues the TDM DMA (EXEC ignored by TDM)
    unsigned long long ga = (unsigned long long)(const void*)bmsg;
    unsigned lds = (unsigned)(unsigned long long)(const void*)&Bsm[0];
    v4u g0;
    g0[0] = 1u;                                            // count=1, user D#
    g0[1] = lds;                                           // lds_addr
    g0[2] = (unsigned)(ga & 0xFFFFFFFFull);                // global_addr lo
    g0[3] = (unsigned)((ga >> 32) & 0x01FFFFFFull) | (2u << 30);  // hi | type=2
    v8u g1;
    g1[0] = 2u << 16;            // workgroup_mask=0, data_size=2 (4 bytes)
    g1[1] = 0x1C00u << 16;       // tensor_dim0 = 7168 dwords (bits 79:48 lo16)
    g1[2] = 1u << 16;            // tensor_dim0 hi=0, tensor_dim1 = 1
    g1[3] = 0x1C00u << 16;       // tile_dim0 = 7168 (bits 127:112)
    g1[4] = 1u;                  // tile_dim1 = 1, tile_dim2 = 0
    g1[5] = 0x1C00u;             // tensor_dim0_stride = 7168
    g1[6] = 0u;
    g1[7] = 0u;
    asm volatile("tensor_load_to_lds %0, %1" :: "s"(g0), "s"(g1) : "memory");
    __builtin_amdgcn_s_wait_tensorcnt(0);
  }
  __syncthreads();

  const int lane = threadIdx.x & 31;
  const int wave = threadIdx.x >> 5;
  const int tile = blockIdx.x * 8 + wave;
  if (tile >= ETILES) return;  // wave-uniform; live waves keep EXEC all-ones

  const int my_e   = tile * 16 + (lane & 15);
  const int my_src = edge_index[my_e];
  const int my_dst = edge_index[N_EDGES + my_e];

  // edge-network inputs for this lane's row; u[13] = 1 feeds the bias matrix
  float uf[DU];
#pragma unroll
  for (int d = 0; d < DE; ++d) uf[d] = edge_attr[my_e * DE + d];
  uf[DE] = 1.0f;

  // gather x[src] K-slice into A-layout order (kept f32 for scale precision)
  const int koff = (lane < 16) ? 0 : 8;
  const float* xr = x + (size_t)my_src * DIN + koff;
  float yf[16];
#pragma unroll
  for (int j = 0; j < 8; ++j) { yf[j] = xr[j]; yf[8 + j] = xr[16 + j]; }

  const v16bf* B = (const v16bf*)Bsm;
  v8f c0 = {}, c1 = {};
#pragma unroll
  for (int d = 0; d < DU; ++d) {
    const float s = uf[d];
    v16bf a;
#pragma unroll
    for (int j = 0; j < 16; ++j) a[j] = (__bf16)(s * yf[j]);
    const v16bf b0 = B[(d * 2 + 0) * 32 + lane];
    const v16bf b1 = B[(d * 2 + 1) * 32 + lane];
    c0 = __builtin_amdgcn_wmma_f32_16x16x32_bf16(false, a, false, b0,
                                                 (short)0, c0, false, false);
    c1 = __builtin_amdgcn_wmma_f32_16x16x32_bf16(false, a, false, b1,
                                                 (short)0, c1, false, false);
  }

  // scatter: C layout reg r = rows {r, r+8}; route dst via cross-lane shuffle
  const int colbase = lane & 15;
  const int halfadd = (lane < 16) ? 0 : 8;
#pragma unroll
  for (int r = 0; r < 8; ++r) {
    const int dd = __shfl(my_dst, r + halfadd, 32);
    atomicAdd(&agg[(size_t)dd * DOUT + colbase], c0[r]);
    atomicAdd(&agg[(size_t)dd * DOUT + 16 + colbase], c1[r]);
  }
  if (lane < 16) atomicAdd(&cnt[my_dst], 1.0f);
}

// ---------------------------------------------------------------------------
// K3: node kernel. mean + x@root + bias -> CELU -> GRU step -> residual ReLU.
// Node weight table (14 KB) staged to LDS with async global->LDS b128 loads
// (ASYNCcnt-tracked, no VGPR round-trip). c moves C-layout -> A-layout via a
// per-wave LDS staging tile (same-wave ds ops are in-order).
// ---------------------------------------------------------------------------
__global__ __launch_bounds__(256) void node_kernel(
    const float* __restrict__ x, const float* __restrict__ bias,
    const float* __restrict__ b_ih, const float* __restrict__ b_hh,
    const __bf16* __restrict__ bnode, const float* __restrict__ agg,
    const float* __restrict__ cnt, float* __restrict__ out,
    float* __restrict__ hnew) {
  __shared__ __align__(32) __bf16 Bsm[14 * 512];
  __shared__ float Stage[8][16 * DOUT];
  {
    const unsigned ldsBase = (unsigned)(unsigned long long)(const void*)&Bsm[0];
    for (int i = threadIdx.x; i < 14 * 512 * 2 / 16; i += 256) {
      unsigned lds  = ldsBase + (unsigned)i * 16u;
      unsigned voff = (unsigned)i * 16u;
      asm volatile("global_load_async_to_lds_b128 %0, %1, %2"
                   :: "v"(lds), "v"(voff), "s"(bnode) : "memory");
    }
    asm volatile("s_wait_asynccnt 0x0" ::: "memory");
  }
  __syncthreads();

  const int lane = threadIdx.x & 31;
  const int wave = threadIdx.x >> 5;
  const int tile = blockIdx.x * 8 + wave;
  if (tile >= NTILES) return;

  const int nbase   = tile * 16;
  const int m       = lane & 15;
  const int koff    = (lane < 16) ? 0 : 8;
  const int halfadd = (lane < 16) ? 0 : 8;
  const v16bf* B = (const v16bf*)Bsm;
  float* stg = Stage[wave];

  // x rows in A layout (also serves as h0 operand for the hidden GEMM)
  const float* xr = x + (size_t)(nbase + m) * DIN + koff;
  v16bf xa;
#pragma unroll
  for (int j = 0; j < 8; ++j) {
    xa[j]     = (__bf16)xr[j];
    xa[8 + j] = (__bf16)xr[16 + j];
  }

  const v8f zacc = {};
  v8f conv0 = __builtin_amdgcn_wmma_f32_16x16x32_bf16(false, xa, false,
                 B[0 * 32 + lane], (short)0, zacc, false, false);
  v8f conv1 = __builtin_amdgcn_wmma_f32_16x16x32_bf16(false, xa, false,
                 B[1 * 32 + lane], (short)0, zacc, false, false);

  // mean aggregation + root + bias, CELU, stage to LDS in row-major f32
#pragma unroll
  for (int t = 0; t < 2; ++t) {
    const int g = t * 16 + (lane & 15);
    const float bv = bias[g];
#pragma unroll
    for (int r = 0; r < 8; ++r) {
      const int nr = nbase + r + halfadd;
      const float cv = cnt[nr];
      const float mean = agg[(size_t)nr * DOUT + g] / fmaxf(cv, 1.0f);
      const float convv = mean + ((t == 0) ? conv0[r] : conv1[r]) + bv;
      const float cel = convv > 0.0f ? convv : (__expf(convv) - 1.0f); // CELU(a=1)
      stg[(r + halfadd) * DOUT + g] = cel;
    }
  }

  // reload c in A layout as bf16 (same-wave LDS ops are in-order)
  const float* srow = stg + m * DOUT;
  v16bf ca;
#pragma unroll
  for (int j = 0; j < 8; ++j) {
    ca[j]     = (__bf16)srow[koff + j];
    ca[8 + j] = (__bf16)srow[koff + 16 + j];
  }

  // GRU gate GEMMs: gi = c @ w_ih^T, gh = h0 @ w_hh^T  (96 outputs = 6 tiles)
  v8f gi[6], gh[6];
#pragma unroll
  for (int t = 0; t < 6; ++t) {
    gi[t] = __builtin_amdgcn_wmma_f32_16x16x32_bf16(false, ca, false,
              B[(2 + t) * 32 + lane], (short)0, zacc, false, false);
    gh[t] = __builtin_amdgcn_wmma_f32_16x16x32_bf16(false, xa, false,
              B[(8 + t) * 32 + lane], (short)0, zacc, false, false);
  }

  // fused gate math + residual ReLU
#pragma unroll
  for (int tc = 0; tc < 2; ++tc) {
    const int g = tc * 16 + (lane & 15);
    const float bi0 = b_ih[g], bi1 = b_ih[32 + g], bi2 = b_ih[64 + g];
    const float bh0 = b_hh[g], bh1 = b_hh[32 + g], bh2 = b_hh[64 + g];
#pragma unroll
    for (int r = 0; r < 8; ++r) {
      const int nr = nbase + r + halfadd;
      const float ir = gi[tc][r] + bi0,     hr = gh[tc][r] + bh0;
      const float iz = gi[2 + tc][r] + bi1, hz = gh[2 + tc][r] + bh1;
      const float ic = gi[4 + tc][r] + bi2, hc = gh[4 + tc][r] + bh2;
      const float rg = 1.0f / (1.0f + __expf(-(ir + hr)));
      const float zg = 1.0f / (1.0f + __expf(-(iz + hz)));
      const float e2 = __expf(2.0f * (ic + rg * hc));
      const float ng = (e2 - 1.0f) / (e2 + 1.0f);   // tanh
      const float h0v = x[(size_t)nr * DOUT + g];
      const float hv = (1.0f - zg) * ng + zg * h0v;
      const size_t o = (size_t)nr * DOUT + g;
      out[o]  = fmaxf(hv + h0v, 0.0f);
      hnew[o] = hv;
    }
  }
}

// ---------------------------------------------------------------------------
extern "C" void kernel_launch(void* const* d_in, const int* in_sizes, int n_in,
                              void* d_out, int out_size, void* d_ws,
                              size_t ws_size, hipStream_t stream) {
  const float* x         = (const float*)d_in[0];
  const float* edge_attr = (const float*)d_in[1];
  const float* nn_w      = (const float*)d_in[2];
  const float* nn_b      = (const float*)d_in[3];
  const float* root      = (const float*)d_in[4];
  const float* bias      = (const float*)d_in[5];
  const float* w_ih      = (const float*)d_in[6];
  const float* w_hh      = (const float*)d_in[7];
  const float* b_ih      = (const float*)d_in[8];
  const float* b_hh      = (const float*)d_in[9];
  const int* edge_index  = (const int*)d_in[10];

  char* ws = (char*)d_ws;
  float*  agg   = (float*)(ws + OFF_AGG);
  float*  cnt   = (float*)(ws + OFF_CNT);
  __bf16* bmsg  = (__bf16*)(ws + OFF_BMSG);
  __bf16* bnode = (__bf16*)(ws + OFF_BNODE);

  float* out  = (float*)d_out;
  float* hnew = out + (size_t)N_NODES * DOUT;

  pack_weights<<<56, 256, 0, stream>>>(nn_w, nn_b, root, w_ih, w_hh, bmsg, bnode);
  zero_ws<<<(N_NODES * (DOUT + 1) + 255) / 256, 256, 0, stream>>>(agg);
  edge_kernel<<<(ETILES + 7) / 8, 256, 0, stream>>>(x, edge_attr, edge_index,
                                                    bmsg, agg, cnt);
  node_kernel<<<(NTILES + 7) / 8, 256, 0, stream>>>(x, bias, b_ih, b_hh, bnode,
                                                    agg, cnt, out, hnew);
}
